// MMatrixLayer_16887811407943
// MI455X (gfx1250) — compile-verified, matrix operand order checked by reading, not run on hardware
//
#include <hip/hip_runtime.h>
#include <math.h>

// ---------------------------------------------------------------------------
// MI455X (gfx1250, wave32) fused implementation of the molecular message-
// passing layer.  GEMMs run on V_WMMA_F32_16X16X32_F16 (f16 in, f32 acc).
// Weights are staged into LDS pre-swizzled into WMMA B-fragment order so a
// fragment load is 2x ds_load_b128 instead of 16x ds_load_u16.
// ---------------------------------------------------------------------------

typedef _Float16 v16h __attribute__((ext_vector_type(16)));
typedef float    v8f  __attribute__((ext_vector_type(8)));
typedef _Float16 h8   __attribute__((ext_vector_type(8)));

#define NBASIS   16
#define FEAT     32      // 2*N_BASIS
#define HIDDEN   64
#define OUTD     128

// fast SiLU: hardware v_rcp_f32 instead of the IEEE division expansion
__device__ __forceinline__ float silu_f(float x) {
    return x * __builtin_amdgcn_rcpf(1.0f + __expf(-x));
}

// --- WMMA fragment loaders -------------------------------------------------
// A-matrix (16xK f16) layout per CDNA5 ISA 7.12.2:
//   lanes 0-15  : row M=lane,    halves = K {kb+0..7,  kb+16..23}
//   lanes 16-31 : row M=lane-16, halves = K {kb+8..15, kb+24..31}
__device__ __forceinline__ v16h load_a_frag(const _Float16* t, int rowstride, int kbase) {
    int lane = threadIdx.x & 31;
    int m    = lane & 15;
    int koff = kbase + ((lane >> 4) << 3);
    h8 lo = *(const h8*)(t + m * rowstride + koff);
    h8 hi = *(const h8*)(t + m * rowstride + koff + 16);
    return __builtin_shufflevector(lo, hi, 0, 1, 2, 3, 4, 5, 6, 7,
                                   8, 9, 10, 11, 12, 13, 14, 15);
}

// Pre-swizzled B fragment: lane's 16 halfs are contiguous (32 bytes).
__device__ __forceinline__ v16h load_b_fragS(const _Float16* fb, int NT, int nt, int kc) {
    int lane = threadIdx.x & 31;
    return *(const v16h*)(fb + (((kc * NT + nt) * 32 + lane) << 4));
}

// Stage a KxN row-major f32 weight matrix into LDS in B-fragment order:
//   dst[((kc*NT + nt)*32 + lane)*16 + i] = w[k][n]  (converted to f16)
// where n = nt*16 + lane%16, koff = (lane>>4)*8,
//       k = kc*32 + (i<8 ? koff+i : 16+koff+(i-8)).
__device__ __forceinline__ void stage_weights_frag(
    const float* __restrict__ w, _Float16* __restrict__ dst,
    int K, int N, int tid, int nthreads) {
    const int NT    = N >> 4;
    const int total = K * N;
    for (int t = tid; t < total; t += nthreads) {
        int i    = t & 15;
        int lane = (t >> 4) & 31;
        int f    = t >> 9;            // fragment index = kc*NT + nt
        int nt   = f % NT;
        int kc   = f / NT;
        int n    = nt * 16 + (lane & 15);
        int koff = (lane >> 4) << 3;
        int k    = kc * 32 + ((i < 8) ? (koff + i) : (16 + koff + (i - 8)));
        dst[t] = (_Float16)w[k * N + n];
    }
}

// --- spherical-harmonics constants ----------------------------------------
#define SH_C0 0.28209479177387814f
#define SH_C1 0.48860251190291992f
#define SH_C2 1.09254843059207907f
#define SH_C3 0.31539156525252005f
#define SH_C4 0.54627421529603953f

// Compute 16 features for this lane's row:
//   role 0 (lanes 0-15):  RBF(d)         -> feature cols 0..15
//   role 1 (lanes 16-31): SH(u) @ sh_w+b -> feature cols 16..31
__device__ __forceinline__ void make_features(
    float rx, float ry, float rz, int role,
    const float* __restrict__ centers, const float* __restrict__ widths,
    const float* __restrict__ sh_w,    const float* __restrict__ sh_b,
    _Float16* fv /* alignas(16) [16] */) {
    float d = sqrtf(rx * rx + ry * ry + rz * rz);
    d = fmaxf(d, 1e-6f);
    if (role == 0) {
#pragma unroll
        for (int j = 0; j < NBASIS; ++j) {
            float t = d - centers[j];
            fv[j] = (_Float16)__expf(-fabsf(widths[j]) * t * t);
        }
    } else {
        float inv = __builtin_amdgcn_rcpf(d);
        float x = rx * inv, y = ry * inv, z = rz * inv;
        float s[9];
        s[0] = SH_C0;
        s[1] = SH_C1 * y;
        s[2] = SH_C1 * z;
        s[3] = SH_C1 * x;
        s[4] = SH_C2 * x * y;
        s[5] = SH_C2 * y * z;
        s[6] = SH_C3 * (2.0f * z * z - x * x - y * y);
        s[7] = SH_C2 * x * z;
        s[8] = SH_C4 * (x * x - y * y);
#pragma unroll
        for (int j = 0; j < NBASIS; ++j) {
            float a = sh_b[j];
#pragma unroll
            for (int i = 0; i < 9; ++i) a += s[i] * sh_w[i * NBASIS + j];
            fv[j] = (_Float16)a;
        }
    }
}

// ---------------------------------------------------------------------------
__global__ void zero_kernel(float* __restrict__ p, int n) {
    int i = blockIdx.x * blockDim.x + threadIdx.x;
    if (i < n) p[i] = 0.0f;
}

// ---------------------------------------------------------------------------
// Edge kernel: one wave32 per 16-edge tile.  8 waves / block.
// ---------------------------------------------------------------------------
__global__ __launch_bounds__(256) void edge_kernel(
    const float* __restrict__ pos, const int* __restrict__ eidx,
    const float* __restrict__ centers, const float* __restrict__ widths,
    const float* __restrict__ sh_w, const float* __restrict__ sh_b,
    const float* __restrict__ w1, const float* __restrict__ b1,
    const float* __restrict__ w2, const float* __restrict__ b2,
    float* __restrict__ agg, int nE) {
    __shared__ __align__(32) _Float16 sW1[FEAT * HIDDEN];     //  4 KB (frag order)
    __shared__ __align__(32) _Float16 sW2[HIDDEN * HIDDEN];   //  8 KB (frag order)
    __shared__ __align__(16) _Float16 sFeat[8][16 * FEAT];    //  8 KB
    __shared__ __align__(16) _Float16 sH1[8][16 * HIDDEN];    // 16 KB
    __shared__ int sSrc[8][16];

    stage_weights_frag(w1, sW1, FEAT,   HIDDEN, threadIdx.x, blockDim.x);
    stage_weights_frag(w2, sW2, HIDDEN, HIDDEN, threadIdx.x, blockDim.x);
    __syncthreads();

    const int wave = threadIdx.x >> 5;
    const int lane = threadIdx.x & 31;
    const int m    = lane & 15;
    const int hiH  = lane >> 4;      // half-wave role
    const int tile = blockIdx.x * 8 + wave;
    if (tile * 16 >= nE) return;

    // --- per-edge geometry + features (16 edges, lane-split roles) ---------
    int e = tile * 16 + m;  if (e >= nE) e = nE - 1;
    int s = eidx[e];
    int d = eidx[nE + e];
    float rx = pos[d * 3 + 0] - pos[s * 3 + 0];
    float ry = pos[d * 3 + 1] - pos[s * 3 + 1];
    float rz = pos[d * 3 + 2] - pos[s * 3 + 2];
    if (hiH == 0) sSrc[wave][m] = s;

    __attribute__((aligned(16))) _Float16 fv[16];
    make_features(rx, ry, rz, hiH, centers, widths, sh_w, sh_b, fv);
    {
        h8* dst = (h8*)(&sFeat[wave][m * FEAT + hiH * 16]);
        dst[0] = *(const h8*)&fv[0];
        dst[1] = *(const h8*)&fv[8];
    }
    // wave-private LDS region: in-wave DS ordering guarantees visibility.

    // --- GEMM1: feat(16x32) @ w1(32x64) + b1, SiLU --------------------------
    v16h a1 = load_a_frag(sFeat[wave], FEAT, 0);
    v8f acc1[4];
#pragma unroll
    for (int nt = 0; nt < 4; ++nt) {
        float bb = b1[nt * 16 + m];
        v8f c;
#pragma unroll
        for (int r = 0; r < 8; ++r) c[r] = bb;
        v16h bf = load_b_fragS(sW1, 4, nt, 0);
        acc1[nt] = __builtin_amdgcn_wmma_f32_16x16x32_f16(false, a1, false, bf,
                                                          (short)0, c, false, false);
    }
#pragma unroll
    for (int nt = 0; nt < 4; ++nt)
#pragma unroll
        for (int r = 0; r < 8; ++r) {
            float v = silu_f(acc1[nt][r]);
            sH1[wave][(r + 8 * hiH) * HIDDEN + nt * 16 + m] = (_Float16)v;
        }

    // --- GEMM2: h1(16x64) @ w2(64x64) + b2, scatter-add ---------------------
    v16h a2lo = load_a_frag(sH1[wave], HIDDEN, 0);
    v16h a2hi = load_a_frag(sH1[wave], HIDDEN, 32);
#pragma unroll
    for (int nt = 0; nt < 4; ++nt) {
        float bb = b2[nt * 16 + m];
        v8f c;
#pragma unroll
        for (int r = 0; r < 8; ++r) c[r] = bb;
        v16h bf0 = load_b_fragS(sW2, 4, nt, 0);
        v16h bf1 = load_b_fragS(sW2, 4, nt, 1);
        c = __builtin_amdgcn_wmma_f32_16x16x32_f16(false, a2lo, false, bf0, (short)0, c, false, false);
        c = __builtin_amdgcn_wmma_f32_16x16x32_f16(false, a2hi, false, bf1, (short)0, c, false, false);
#pragma unroll
        for (int r = 0; r < 8; ++r) {
            int row = r + 8 * hiH;
            int ee  = tile * 16 + row;
            if (ee < nE)
                atomicAdd(&agg[(size_t)sSrc[wave][row] * HIDDEN + nt * 16 + m], c[r]);
        }
    }
}

// ---------------------------------------------------------------------------
// Node kernel: one wave32 per 16-node tile.  4 waves / block (56 KB LDS).
// ---------------------------------------------------------------------------
__global__ __launch_bounds__(128) void node_kernel(
    const float* __restrict__ pos, const float* __restrict__ zpos,
    const float* __restrict__ centers, const float* __restrict__ widths,
    const float* __restrict__ sh_w, const float* __restrict__ sh_b,
    const float* __restrict__ znw, const float* __restrict__ znb,
    const float* __restrict__ npw, const float* __restrict__ npb,
    const float* __restrict__ lng, const float* __restrict__ lnb,
    const float* __restrict__ agg, float* __restrict__ out, int nN) {
    __shared__ __align__(32) _Float16 sZn[FEAT * HIDDEN];    //  4 KB (frag order)
    __shared__ __align__(32) _Float16 sNp[OUTD * OUTD];      // 32 KB (frag order)
    __shared__ __align__(16) _Float16 sFeat[4][16 * FEAT];   //  4 KB
    __shared__ __align__(16) _Float16 sA2[4][16 * OUTD];     // 16 KB

    stage_weights_frag(znw, sZn, FEAT, HIDDEN, threadIdx.x, blockDim.x);
    stage_weights_frag(npw, sNp, OUTD, OUTD,   threadIdx.x, blockDim.x);
    __syncthreads();

    const int wave = threadIdx.x >> 5;
    const int lane = threadIdx.x & 31;
    const int m    = lane & 15;
    const int hiH  = lane >> 4;
    const int tile = blockIdx.x * 4 + wave;
    if (tile * 16 >= nN) return;

    // --- zinc features ------------------------------------------------------
    int node = tile * 16 + m;  if (node >= nN) node = nN - 1;
    float rx = pos[node * 3 + 0] - zpos[0];
    float ry = pos[node * 3 + 1] - zpos[1];
    float rz = pos[node * 3 + 2] - zpos[2];
    __attribute__((aligned(16))) _Float16 fv[16];
    make_features(rx, ry, rz, hiH, centers, widths, sh_w, sh_b, fv);
    {
        h8* dst = (h8*)(&sFeat[wave][m * FEAT + hiH * 16]);
        dst[0] = *(const h8*)&fv[0];
        dst[1] = *(const h8*)&fv[8];
    }

    // --- zn MLP: zfeat(16x32) @ zn_w(32x64) + b, SiLU -> sA2 cols 64..127 ---
    v16h az = load_a_frag(sFeat[wave], FEAT, 0);
#pragma unroll
    for (int nt = 0; nt < 4; ++nt) {
        float bb = znb[nt * 16 + m];
        v8f c;
#pragma unroll
        for (int r = 0; r < 8; ++r) c[r] = bb;
        v16h bf = load_b_fragS(sZn, 4, nt, 0);
        c = __builtin_amdgcn_wmma_f32_16x16x32_f16(false, az, false, bf, (short)0, c, false, false);
#pragma unroll
        for (int r = 0; r < 8; ++r) {
            float v = silu_f(c[r]);
            sA2[wave][(r + 8 * hiH) * OUTD + HIDDEN + nt * 16 + m] = (_Float16)v;
        }
    }

    // --- agg rows -> sA2 cols 0..63 (each lane: half a row, 32 f32 -> f16) --
    {
        int row = m;
        int cb  = hiH * 32;
        int rr  = tile * 16 + row;  if (rr >= nN) rr = nN - 1;
        const float* g = agg + (size_t)rr * HIDDEN + cb;
        __attribute__((aligned(16))) _Float16 tmp[32];
#pragma unroll
        for (int j = 0; j < 32; ++j) tmp[j] = (_Float16)g[j];
        h8* dst = (h8*)(&sA2[wave][row * OUTD + cb]);
#pragma unroll
        for (int q = 0; q < 4; ++q) dst[q] = *(const h8*)&tmp[q * 8];
    }

    // --- big GEMM: [agg,zn](16x128) @ np_w(128x128) + np_b ------------------
    v16h af[4];
#pragma unroll
    for (int kc = 0; kc < 4; ++kc) af[kc] = load_a_frag(sA2[wave], OUTD, kc * 32);

    v8f accN[8];
#pragma unroll
    for (int nt = 0; nt < 8; ++nt) {
        float bb = npb[nt * 16 + m];
        v8f c;
#pragma unroll
        for (int r = 0; r < 8; ++r) c[r] = bb;
#pragma unroll
        for (int kc = 0; kc < 4; ++kc) {
            v16h bf = load_b_fragS(sNp, 8, nt, kc);
            c = __builtin_amdgcn_wmma_f32_16x16x32_f16(false, af[kc], false, bf,
                                                       (short)0, c, false, false);
        }
        accN[nt] = c;
    }

    // --- LayerNorm (rows reduced across half-wave via shfl_xor) + SiLU ------
    float gg[8], bv[8];
#pragma unroll
    for (int nt = 0; nt < 8; ++nt) {
        gg[nt] = lng[nt * 16 + m];
        bv[nt] = lnb[nt * 16 + m];
    }
#pragma unroll
    for (int r = 0; r < 8; ++r) {
        float s1 = 0.0f, s2 = 0.0f;
#pragma unroll
        for (int nt = 0; nt < 8; ++nt) {
            float v = accN[nt][r];
            s1 += v; s2 += v * v;
        }
        // reduce across the 16 lanes of this half-wave (masks < 16 stay in-half)
#pragma unroll
        for (int msk = 1; msk < 16; msk <<= 1) {
            s1 += __shfl_xor(s1, msk, 32);
            s2 += __shfl_xor(s2, msk, 32);
        }
        float mean = s1 * (1.0f / 128.0f);
        float var  = s2 * (1.0f / 128.0f) - mean * mean;
        float rstd = rsqrtf(var + 1e-5f);
        int row = tile * 16 + r + 8 * hiH;
        if (row < nN) {
#pragma unroll
            for (int nt = 0; nt < 8; ++nt) {
                float y = (accN[nt][r] - mean) * rstd * gg[nt] + bv[nt];
                out[(size_t)row * OUTD + nt * 16 + m] = silu_f(y);
            }
        }
    }
}

// ---------------------------------------------------------------------------
extern "C" void kernel_launch(void* const* d_in, const int* in_sizes, int n_in,
                              void* d_out, int out_size, void* d_ws, size_t ws_size,
                              hipStream_t stream) {
    const float* pos      = (const float*)d_in[0];
    const float* zpos     = (const float*)d_in[1];
    const int*   eidx     = (const int*)d_in[2];
    const float* centers  = (const float*)d_in[3];
    const float* widths   = (const float*)d_in[4];
    const float* sh_w     = (const float*)d_in[5];
    const float* sh_b     = (const float*)d_in[6];
    const float* pm_w1    = (const float*)d_in[7];
    const float* pm_b1    = (const float*)d_in[8];
    const float* pm_w2    = (const float*)d_in[9];
    const float* pm_b2    = (const float*)d_in[10];
    const float* zn_w     = (const float*)d_in[11];
    const float* zn_b     = (const float*)d_in[12];
    const float* np_w     = (const float*)d_in[13];
    const float* np_b     = (const float*)d_in[14];
    const float* ln_g     = (const float*)d_in[15];
    const float* ln_b     = (const float*)d_in[16];
    float* out = (float*)d_out;

    const int nN = in_sizes[0] / 3;
    const int nE = in_sizes[2] / 2;

    float* agg = (float*)d_ws;             // nN x 64 f32 accumulator
    const int aggN = nN * HIDDEN;

    zero_kernel<<<(aggN + 255) / 256, 256, 0, stream>>>(agg, aggN);

    const int etiles = (nE + 15) / 16;
    edge_kernel<<<(etiles + 7) / 8, 256, 0, stream>>>(
        pos, eidx, centers, widths, sh_w, sh_b,
        pm_w1, pm_b1, pm_w2, pm_b2, agg, nE);

    const int ntiles = (nN + 15) / 16;
    node_kernel<<<(ntiles + 3) / 4, 128, 0, stream>>>(
        pos, zpos, centers, widths, sh_w, sh_b,
        zn_w, zn_b, np_w, np_b, ln_g, ln_b, agg, out, nN);
}
